// C2M_24378234372461
// MI455X (gfx1250) — matrix operand order repaired and checked by reading, not compiled, not run on hardware
//
#include <hip/hip_runtime.h>
#include <hip/hip_bf16.h>
#include <math.h>

// ======================================================================
// CDNA5 (gfx1250, wave32) implementation.
// All GEMMs / convs are implicit-GEMM on v_wmma_f32_16x16x32_f16.
// Block = 128 threads = 4 waves; block tile 64(M) x 64(N); each wave owns
// a 32x32 sub-tile = 2x2 WMMA fragments (4 v_wmma per K-step).
// Software pipeline: double-buffered LDS (ping-pong), global loads for
// chunk k+1 issued before the WMMAs of chunk k, ONE barrier per K-step.
// LDS tile writes are explicit 8xf16 vectors -> ds_store_b128.
// ======================================================================

typedef __attribute__((ext_vector_type(16))) _Float16 v16h;
typedef __attribute__((ext_vector_type(8)))  _Float16 v8h;
typedef __attribute__((ext_vector_type(8)))  float    v8f;

__device__ __forceinline__ v8f wmma16(v16h a, v16h b, v8f c) {
  return __builtin_amdgcn_wmma_f32_16x16x32_f16(false, a, false, b,
                                                (short)0, c, false, false);
}

// A fragment: 16x32 (MxK) f16. lane<16: row=lane, K 0..15;
// lane>=16: row=lane-16, K 16..31. As rows are 32 halves.
__device__ __forceinline__ v16h frag_a_lds(const _Float16* __restrict__ As, int lane) {
  const int row = lane & 15;
  const int kb  = (lane >> 4) << 4;          // 0 or 16
  const _Float16* p = As + row * 32 + kb;
  v16h f;
#pragma unroll
  for (int i = 0; i < 16; ++i) f[i] = p[i];
  return f;
}

// B fragment: 32x16 (KxN) f16. lane<16: col=lane, K 0..15;
// lane>=16: col=lane-16, K 16..31. Bt stored n-major: Bt[n][k], 32 halves/row.
__device__ __forceinline__ v16h frag_b_lds(const _Float16* __restrict__ Bt, int lane) {
  const int col = lane & 15;
  const int kb  = (lane >> 4) << 4;          // 0 or 16
  const _Float16* p = Bt + col * 32 + kb;
  v16h f;
#pragma unroll
  for (int i = 0; i < 16; ++i) f[i] = p[i];
  return f;
}

// pack 16 staged floats into two v8h and write 2x ds_store_b128
__device__ __forceinline__ void lds_store16(_Float16* dst, const float* r) {
  v8h h0, h1;
#pragma unroll
  for (int j = 0; j < 8; ++j) { h0[j] = (_Float16)r[j]; h1[j] = (_Float16)r[j + 8]; }
  v8h* d = (v8h*)dst;           // 16B aligned: base align(16), offset l*64B + {0,32}B
  d[0] = h0;
  d[1] = h1;
}

// ---------------------------------------------------------------------
// Implicit-GEMM convolution (NCHW, OIHW weights); geometry compile-time.
// M=Cout (runtime; must be multiple of 64), N=OUTH*OUTW, K=CIN*KH*KW.
// ---------------------------------------------------------------------
template <int CIN, int H, int W, int KH, int KW, int PADH, int PADW,
          int OUTH, int OUTW>
__global__ __launch_bounds__(128) void conv_gemm_wmma(
    const float* __restrict__ X, const float* __restrict__ Wt,
    const float* __restrict__ bias, float* __restrict__ Out, int Cout) {
  constexpr int N  = OUTH * OUTW;
  constexpr int KK = CIN * KH * KW;
  constexpr int KHW = KH * KW;
  static_assert(KK % 32 == 0, "K must be a multiple of 32");

  __shared__ __align__(16) _Float16 As[2][64 * 32];
  __shared__ __align__(16) _Float16 Bt[2][64 * 32];

  const int b = blockIdx.z;
  const float* Xb = X + (long)b * CIN * H * W;
  float* Ob = Out + (long)b * Cout * N;

  const int m0 = blockIdx.y * 64;
  const int n0 = blockIdx.x * 64;
  const int tid  = threadIdx.x;
  const int lane = tid & 31;
  const int wid  = tid >> 5;
  const int wm   = wid >> 1;
  const int wn   = wid & 1;

  // staging geometry: thread owns row l (0..63), contiguous 16-wide K chunk
  const int l  = tid & 63;
  const int kb = (tid >> 6) << 4;            // 0 or 16
  const int mA = m0 + l;                     // always < Cout (Cout % 64 == 0)
  const int p  = n0 + l;                     // pixel, may exceed N on edge
  const int oy = p / OUTW;                   // constant divisor
  const int ox = p - oy * OUTW;
  const bool pok = (p < N);

  v8f a00 = {}, a01 = {}, a10 = {}, a11 = {};

  float ra[16], rb[16];

  auto load_chunk = [&](int k0) {
    // A tile: weights, 16 contiguous floats, provably 16B aligned
    const float4* pa4 = (const float4*)(Wt + (long)mA * KK + (k0 + kb));
#pragma unroll
    for (int j = 0; j < 4; ++j) {
      const float4 w = pa4[j];
      ra[4 * j + 0] = w.x; ra[4 * j + 1] = w.y;
      ra[4 * j + 2] = w.z; ra[4 * j + 3] = w.w;
    }
    // B tile (im2col), branchless selects
#pragma unroll
    for (int j = 0; j < 16; ++j) {
      const int k  = k0 + kb + j;
      const int ci = k / KHW;
      const int rr = k - ci * KHW;
      const int ky = rr / KW;
      const int kx = rr - ky * KW;
      const int y  = oy - PADH + ky;
      const int x  = ox - PADW + kx;
      const bool ok = pok & (y >= 0) & (y < H) & (x >= 0) & (x < W);
      const float v = Xb[ok ? ((ci * H + y) * W + x) : 0];
      rb[j] = ok ? v : 0.f;
    }
  };
  auto store_chunk = [&](int buf) {
    lds_store16(As[buf] + l * 32 + kb, ra);
    lds_store16(Bt[buf] + l * 32 + kb, rb);
  };
  auto mma_step = [&](int buf) {
    v16h fa0 = frag_a_lds(As[buf] + (wm * 32 +  0) * 32, lane);
    v16h fa1 = frag_a_lds(As[buf] + (wm * 32 + 16) * 32, lane);
    v16h fb0 = frag_b_lds(Bt[buf] + (wn * 32 +  0) * 32, lane);
    v16h fb1 = frag_b_lds(Bt[buf] + (wn * 32 + 16) * 32, lane);
    a00 = wmma16(fa0, fb0, a00);
    a01 = wmma16(fa0, fb1, a01);
    a10 = wmma16(fa1, fb0, a10);
    a11 = wmma16(fa1, fb1, a11);
  };

  load_chunk(0);
  store_chunk(0);
  __syncthreads();
  int cur = 0;
  for (int k0 = 32; k0 < KK; k0 += 32) {
    load_chunk(k0);           // global loads for next chunk (in flight)
    mma_step(cur);            // compute current chunk
    store_chunk(cur ^ 1);     // write next chunk to the other buffer
    __syncthreads();
    cur ^= 1;
  }
  mma_step(cur);

  const int mo = (lane >> 4) * 8;
  const int nn = lane & 15;
#define C2M_STORE_CONV(ACC, MI, NI)                                         \
  {                                                                         \
    const int bm = m0 + wm * 32 + (MI)*16 + mo;                             \
    const int n  = n0 + wn * 32 + (NI)*16 + nn;                             \
    if (n < N) {                                                            \
      _Pragma("unroll") for (int j = 0; j < 8; ++j) {                       \
        const int m = bm + j;                                               \
        Ob[(long)m * N + n] = ACC[j] + bias[m];                             \
      }                                                                     \
    }                                                                       \
  }
  C2M_STORE_CONV(a00, 0, 0)
  C2M_STORE_CONV(a01, 0, 1)
  C2M_STORE_CONV(a10, 1, 0)
  C2M_STORE_CONV(a11, 1, 1)
#undef C2M_STORE_CONV
}

// ---------------------------------------------------------------------
// Scores GEMM: Out[M,N] = A^T * B ; A:[K,M], B:[K,N]. Batched (blockIdx.z).
// ---------------------------------------------------------------------
__global__ __launch_bounds__(128) void gemm_tn_wmma(
    const float* __restrict__ A, const float* __restrict__ B,
    float* __restrict__ Out, int M, int N, int K, long sA, long sB, long sO) {
  __shared__ __align__(16) _Float16 As[2][64 * 32];
  __shared__ __align__(16) _Float16 Bt[2][64 * 32];

  const int b = blockIdx.z;
  const float* Ab = A + (long)b * sA;
  const float* Bb = B + (long)b * sB;
  float* Ob = Out + (long)b * sO;

  const int m0 = blockIdx.y * 64;
  const int n0 = blockIdx.x * 64;
  const int tid  = threadIdx.x;
  const int lane = tid & 31;
  const int wid  = tid >> 5;
  const int wm   = wid >> 1;
  const int wn   = wid & 1;

  const int l  = tid & 63;
  const int kb = (tid >> 6) << 4;
  const int mA = m0 + l;
  const int nB = n0 + l;
  const bool mok = (mA < M);
  const bool nok = (nB < N);
  const bool fastMN = (m0 + 64 <= M) && (n0 + 64 <= N);

  v8f a00 = {}, a01 = {}, a10 = {}, a11 = {};
  float ra[16], rb[16];

  auto load_chunk = [&](int k0) {
    if (fastMN && (k0 + 32 <= K)) {
      const float* pa = Ab + (long)(k0 + kb) * M + mA;   // stride M per j
      const float* pb = Bb + (long)(k0 + kb) * N + nB;   // stride N per j
#pragma unroll
      for (int j = 0; j < 16; ++j) { ra[j] = pa[j * M]; rb[j] = pb[j * N]; }
      if (k0 + 32 < K)   // next-chunk hint -> global_prefetch_b8
        __builtin_prefetch(pb + 32 * (long)N, 0, 0);
    } else {
#pragma unroll
      for (int j = 0; j < 16; ++j) {
        const int k = k0 + kb + j;
        const bool kok = (k < K);
        const bool aok = mok & kok;
        const bool bok = nok & kok;
        const float va = Ab[aok ? ((long)k * M + mA) : 0];
        const float vb = Bb[bok ? ((long)k * N + nB) : 0];
        ra[j] = aok ? va : 0.f;
        rb[j] = bok ? vb : 0.f;
      }
    }
  };
  auto store_chunk = [&](int buf) {
    lds_store16(As[buf] + l * 32 + kb, ra);
    lds_store16(Bt[buf] + l * 32 + kb, rb);
  };
  auto mma_step = [&](int buf) {
    v16h fa0 = frag_a_lds(As[buf] + (wm * 32 +  0) * 32, lane);
    v16h fa1 = frag_a_lds(As[buf] + (wm * 32 + 16) * 32, lane);
    v16h fb0 = frag_b_lds(Bt[buf] + (wn * 32 +  0) * 32, lane);
    v16h fb1 = frag_b_lds(Bt[buf] + (wn * 32 + 16) * 32, lane);
    a00 = wmma16(fa0, fb0, a00);
    a01 = wmma16(fa0, fb1, a01);
    a10 = wmma16(fa1, fb0, a10);
    a11 = wmma16(fa1, fb1, a11);
  };

  load_chunk(0);
  store_chunk(0);
  __syncthreads();
  int cur = 0;
  for (int k0 = 32; k0 < K; k0 += 32) {
    load_chunk(k0);
    mma_step(cur);
    store_chunk(cur ^ 1);
    __syncthreads();
    cur ^= 1;
  }
  mma_step(cur);

  const int mo = (lane >> 4) * 8;
  const int nn = lane & 15;
#define C2M_STORE_TN(ACC, MI, NI)                                           \
  {                                                                         \
    const int bm = m0 + wm * 32 + (MI)*16 + mo;                             \
    const int n  = n0 + wn * 32 + (NI)*16 + nn;                             \
    if (n < N) {                                                            \
      _Pragma("unroll") for (int j = 0; j < 8; ++j) {                       \
        const int m = bm + j;                                               \
        if (m < M) Ob[(long)m * N + n] = ACC[j];                            \
      }                                                                     \
    }                                                                       \
  }
  C2M_STORE_TN(a00, 0, 0)
  C2M_STORE_TN(a01, 0, 1)
  C2M_STORE_TN(a10, 1, 0)
  C2M_STORE_TN(a11, 1, 1)
#undef C2M_STORE_TN
}

// ---------------------------------------------------------------------
// Out[M,N] = A * B (+ Add) ; A:[M,K], B:[K,N] row-major. Fused residual.
// ---------------------------------------------------------------------
__global__ __launch_bounds__(128) void gemm_nn_wmma(
    const float* __restrict__ A, const float* __restrict__ B,
    const float* __restrict__ Add, float* __restrict__ Out,
    int M, int N, int K, long sA, long sB, long sAdd, long sO) {
  __shared__ __align__(16) _Float16 As[2][64 * 32];
  __shared__ __align__(16) _Float16 Bt[2][64 * 32];

  const int b = blockIdx.z;
  const float* Ab = A + (long)b * sA;
  const float* Bb = B + (long)b * sB;
  float* Ob = Out + (long)b * sO;

  const int m0 = blockIdx.y * 64;
  const int n0 = blockIdx.x * 64;
  const int tid  = threadIdx.x;
  const int lane = tid & 31;
  const int wid  = tid >> 5;
  const int wm   = wid >> 1;
  const int wn   = wid & 1;

  const int l  = tid & 63;
  const int kb = (tid >> 6) << 4;
  const int mA = m0 + l;
  const int nB = n0 + l;
  const bool mok = (mA < M);
  const bool nok = (nB < N);
  const bool fastMN = (m0 + 64 <= M) && (n0 + 64 <= N);

  v8f a00 = {}, a01 = {}, a10 = {}, a11 = {};
  float ra[16], rb[16];

  auto load_chunk = [&](int k0) {
    if (fastMN && (k0 + 32 <= K)) {
      const float* pa = Ab + (long)mA * K + (k0 + kb);   // 16 contiguous
      const float* pb = Bb + (long)(k0 + kb) * N + nB;   // stride N per j
#pragma unroll
      for (int j = 0; j < 16; ++j) { ra[j] = pa[j]; rb[j] = pb[j * N]; }
      if (k0 + 32 < K)
        __builtin_prefetch(pb + 32 * (long)N, 0, 0);
    } else {
#pragma unroll
      for (int j = 0; j < 16; ++j) {
        const int k = k0 + kb + j;
        const bool kok = (k < K);
        const bool aok = mok & kok;
        const bool bok = nok & kok;
        const float va = Ab[aok ? ((long)mA * K + k) : 0];
        const float vb = Bb[bok ? ((long)k * N + nB) : 0];
        ra[j] = aok ? va : 0.f;
        rb[j] = bok ? vb : 0.f;
      }
    }
  };
  auto store_chunk = [&](int buf) {
    lds_store16(As[buf] + l * 32 + kb, ra);
    lds_store16(Bt[buf] + l * 32 + kb, rb);
  };
  auto mma_step = [&](int buf) {
    v16h fa0 = frag_a_lds(As[buf] + (wm * 32 +  0) * 32, lane);
    v16h fa1 = frag_a_lds(As[buf] + (wm * 32 + 16) * 32, lane);
    v16h fb0 = frag_b_lds(Bt[buf] + (wn * 32 +  0) * 32, lane);
    v16h fb1 = frag_b_lds(Bt[buf] + (wn * 32 + 16) * 32, lane);
    a00 = wmma16(fa0, fb0, a00);
    a01 = wmma16(fa0, fb1, a01);
    a10 = wmma16(fa1, fb0, a10);
    a11 = wmma16(fa1, fb1, a11);
  };

  load_chunk(0);
  store_chunk(0);
  __syncthreads();
  int cur = 0;
  for (int k0 = 32; k0 < K; k0 += 32) {
    load_chunk(k0);
    mma_step(cur);
    store_chunk(cur ^ 1);
    __syncthreads();
    cur ^= 1;
  }
  mma_step(cur);

  const int mo = (lane >> 4) * 8;
  const int nn = lane & 15;
#define C2M_STORE_NN(ACC, MI, NI)                                           \
  {                                                                         \
    const int bm = m0 + wm * 32 + (MI)*16 + mo;                             \
    const int n  = n0 + wn * 32 + (NI)*16 + nn;                             \
    if (n < N) {                                                            \
      _Pragma("unroll") for (int j = 0; j < 8; ++j) {                       \
        const int m = bm + j;                                               \
        if (m < M) {                                                        \
          float v = ACC[j];                                                 \
          if (Add) v += Add[(long)b * sAdd + (long)m * N + n];              \
          Ob[(long)m * N + n] = v;                                          \
        }                                                                   \
      }                                                                     \
    }                                                                       \
  }
  C2M_STORE_NN(a00, 0, 0)
  C2M_STORE_NN(a01, 0, 1)
  C2M_STORE_NN(a10, 1, 0)
  C2M_STORE_NN(a11, 1, 1)
#undef C2M_STORE_NN
}

// ---------------------------------------------------------------------
// Row softmax (in place). One block per row.
// ---------------------------------------------------------------------
__global__ void softmax_rows(float* __restrict__ S, int cols) {
  float* p = S + (long)blockIdx.x * cols;
  const int tid = threadIdx.x;
  __shared__ float red[256];

  float mx = -3.4e38f;
  for (int i = tid; i < cols; i += 256) mx = fmaxf(mx, p[i]);
  red[tid] = mx; __syncthreads();
  for (int st = 128; st > 0; st >>= 1) {
    if (tid < st) red[tid] = fmaxf(red[tid], red[tid + st]);
    __syncthreads();
  }
  mx = red[0]; __syncthreads();

  float s = 0.f;
  for (int i = tid; i < cols; i += 256) {
    const float e = expf(p[i] - mx);
    p[i] = e; s += e;
  }
  red[tid] = s; __syncthreads();
  for (int st = 128; st > 0; st >>= 1) {
    if (tid < st) red[tid] += red[tid + st];
    __syncthreads();
  }
  const float inv = 1.0f / red[0];
  for (int i = tid; i < cols; i += 256) p[i] *= inv;
}

// ---------------------------------------------------------------------
// BatchNorm (training-mode batch stats over (B,H,W)) + ReLU (+residual).
// ---------------------------------------------------------------------
__global__ void bn_stats(const float* __restrict__ X, float* __restrict__ mv,
                         int C, int Bn, int HW) {
  const int c = blockIdx.x;
  const int tid = threadIdx.x;
  const int total = Bn * HW;
  float s = 0.f, q = 0.f;
  for (int i = tid; i < total; i += 256) {
    const int b = i / HW;
    const int r = i - b * HW;
    const float v = X[(long)b * C * HW + (long)c * HW + r];
    s += v; q += v * v;
  }
  __shared__ float rs[256], rq[256];
  rs[tid] = s; rq[tid] = q; __syncthreads();
  for (int st = 128; st > 0; st >>= 1) {
    if (tid < st) { rs[tid] += rs[tid + st]; rq[tid] += rq[tid + st]; }
    __syncthreads();
  }
  if (tid == 0) {
    const float m = rs[0] / total;
    mv[c]     = m;
    mv[C + c] = rq[0] / total - m * m;   // biased variance
  }
}

__global__ void bn_relu_apply(const float* __restrict__ X,
                              const float* __restrict__ mv,
                              const float* __restrict__ g,
                              const float* __restrict__ be,
                              const float* __restrict__ resid,
                              float* __restrict__ Out,
                              int C, int HW, long total) {
  const long i = (long)blockIdx.x * blockDim.x + threadIdx.x;
  if (i >= total) return;
  const int c = (int)((i / HW) % C);
  const float m = mv[c], v = mv[C + c];
  float y = (X[i] - m) * rsqrtf(v + 1e-5f) * g[c] + be[c];
  y = fmaxf(y, 0.f);
  if (resid) y += resid[i];
  Out[i] = y;
}

// ======================================================================
extern "C" void kernel_launch(void* const* d_in, const int* in_sizes, int n_in,
                              void* d_out, int out_size, void* d_ws, size_t ws_size,
                              hipStream_t stream) {
  (void)in_sizes; (void)n_in; (void)out_size; (void)ws_size;

  const float* x2    = (const float*)d_in[0];
  const float* x3    = (const float*)d_in[1];
  const float* x4    = (const float*)d_in[2];
  const float* w_r4q = (const float*)d_in[3];
  const float* b_r4q = (const float*)d_in[4];
  const float* g_r4q = (const float*)d_in[5];
  const float* be_r4q= (const float*)d_in[6];
  const float* w_r4k = (const float*)d_in[7];
  const float* b_r4k = (const float*)d_in[8];
  const float* g_r4k = (const float*)d_in[9];
  const float* be_r4k= (const float*)d_in[10];
  const float* w_n3  = (const float*)d_in[11];
  const float* b_n3  = (const float*)d_in[12];
  const float* w_n2  = (const float*)d_in[13];
  const float* b_n2  = (const float*)d_in[14];
  const float* w1_3q = (const float*)d_in[15];
  const float* b1_3q = (const float*)d_in[16];
  const float* w1_3k = (const float*)d_in[17];
  const float* b1_3k = (const float*)d_in[18];
  const float* w1_2q = (const float*)d_in[19];
  const float* b1_2q = (const float*)d_in[20];
  const float* w1_2k = (const float*)d_in[21];
  const float* b1_2k = (const float*)d_in[22];
  const float* w_2r  = (const float*)d_in[23];
  const float* b_2r  = (const float*)d_in[24];
  const float* g_2r  = (const float*)d_in[25];
  const float* be_2r = (const float*)d_in[26];
  float* out = (float*)d_out;

  const int Bn = 8;
  const int C2 = 128, H2 = 44, W2 = 44, HW2 = H2 * W2;        // 1936
  const int C3 = 256, H3 = 22, W3 = 22, HW3 = H3 * W3;        // 484
  const int HW4 = 121;                                        // 11x11
  const int HW2b = HW2 - 2;                                   // 1934 (k=5, pad=1)

  // ---- workspace layout (floats); att22 aliases the dead phase-1 region ----
  float* ws = (float*)d_ws;
  size_t off = 0;
  float* r2    = ws + off; off += (size_t)Bn * C2 * HW2;
  float* q2    = ws + off; off += (size_t)Bn * C2 * HW2b;
  float* k2    = ws + off; off += (size_t)Bn * C2 * HW2b;
  float* r2f   = ws + off; off += (size_t)Bn * C2 * HW2;
  float* cvo   = ws + off; off += (size_t)Bn * C2 * HW2;
  float* stats = ws + off; off += 2 * (size_t)C3;
  const size_t phase = off;
  float* r4q   = ws + off; off += (size_t)Bn * C3 * HW4;
  float* r4k   = ws + off; off += (size_t)Bn * C3 * HW4;
  float* r3    = ws + off; off += (size_t)Bn * C3 * HW3;
  float* att43 = ws + off; off += (size_t)Bn * HW4 * HW3;
  float* s3    = ws + off; off += (size_t)Bn * C3 * HW3;
  float* q3    = ws + off; off += (size_t)Bn * C2 * HW3;
  float* k3    = ws + off; off += (size_t)Bn * C2 * HW3;
  float* att32 = ws + off; off += (size_t)Bn * HW3 * HW2;
  float* s2    = ws + off; off += (size_t)Bn * C2 * HW2;
  float* att22 = ws + phase;   // alias: phase-1 buffers all dead before this write

  dim3 blk(128, 1, 1);

  // ---- scale-4 Q/K: conv3x3(x4) -> BN -> ReLU ----
  {
    dim3 g((HW4 + 63) / 64, C3 / 64, Bn);
    conv_gemm_wmma<512, 11, 11, 3, 3, 1, 1, 11, 11>
        <<<g, blk, 0, stream>>>(x4, w_r4q, b_r4q, r4q, C3);
    bn_stats<<<dim3(C3), dim3(256), 0, stream>>>(r4q, stats, C3, Bn, HW4);
    long tot = (long)Bn * C3 * HW4;
    bn_relu_apply<<<dim3((unsigned)((tot + 255) / 256)), dim3(256), 0, stream>>>(
        r4q, stats, g_r4q, be_r4q, nullptr, r4q, C3, HW4, tot);

    conv_gemm_wmma<512, 11, 11, 3, 3, 1, 1, 11, 11>
        <<<g, blk, 0, stream>>>(x4, w_r4k, b_r4k, r4k, C3);
    bn_stats<<<dim3(C3), dim3(256), 0, stream>>>(r4k, stats, C3, Bn, HW4);
    bn_relu_apply<<<dim3((unsigned)((tot + 255) / 256)), dim3(256), 0, stream>>>(
        r4k, stats, g_r4k, be_r4k, nullptr, r4k, C3, HW4, tot);
  }

  // ---- r3 = conv3x3(x3) ----
  {
    dim3 g((HW3 + 63) / 64, C3 / 64, Bn);
    conv_gemm_wmma<256, 22, 22, 3, 3, 1, 1, 22, 22>
        <<<g, blk, 0, stream>>>(x3, w_n3, b_n3, r3, C3);
  }

  // ---- att43 = softmax(r4q^T @ r3); s3 = r4k @ att43 + r3 ----
  {
    dim3 g((HW3 + 63) / 64, (HW4 + 63) / 64, Bn);
    gemm_tn_wmma<<<g, blk, 0, stream>>>(r4q, r3, att43, HW4, HW3, C3,
                                        (long)C3 * HW4, (long)C3 * HW3, (long)HW4 * HW3);
    softmax_rows<<<dim3(Bn * HW4), dim3(256), 0, stream>>>(att43, HW3);
    dim3 g2((HW3 + 63) / 64, C3 / 64, Bn);
    gemm_nn_wmma<<<g2, blk, 0, stream>>>(r4k, att43, r3, s3, C3, HW3, HW4,
                                         (long)C3 * HW4, (long)HW4 * HW3,
                                         (long)C3 * HW3, (long)C3 * HW3);
  }

  // ---- q3/k3 = conv1d(s3, k=3, pad=1) ----
  {
    dim3 g((HW3 + 63) / 64, C2 / 64, Bn);
    conv_gemm_wmma<256, 1, 484, 1, 3, 0, 1, 1, 484>
        <<<g, blk, 0, stream>>>(s3, w1_3q, b1_3q, q3, C2);
    conv_gemm_wmma<256, 1, 484, 1, 3, 0, 1, 1, 484>
        <<<g, blk, 0, stream>>>(s3, w1_3k, b1_3k, k3, C2);
  }

  // ---- r2 = conv3x3(x2) ----
  {
    dim3 g((HW2 + 63) / 64, C2 / 64, Bn);
    conv_gemm_wmma<128, 44, 44, 3, 3, 1, 1, 44, 44>
        <<<g, blk, 0, stream>>>(x2, w_n2, b_n2, r2, C2);
  }

  // ---- att32 = softmax(q3^T @ r2); s2 = k3 @ att32 + r2 ----
  {
    dim3 g((HW2 + 63) / 64, (HW3 + 63) / 64, Bn);
    gemm_tn_wmma<<<g, blk, 0, stream>>>(q3, r2, att32, HW3, HW2, C2,
                                        (long)C2 * HW3, (long)C2 * HW2, (long)HW3 * HW2);
    softmax_rows<<<dim3(Bn * HW3), dim3(256), 0, stream>>>(att32, HW2);
    dim3 g2((HW2 + 63) / 64, C2 / 64, Bn);
    gemm_nn_wmma<<<g2, blk, 0, stream>>>(k3, att32, r2, s2, C2, HW2, HW3,
                                         (long)C2 * HW3, (long)HW3 * HW2,
                                         (long)C2 * HW2, (long)C2 * HW2);
  }

  // ---- q2/k2 = conv1d(s2, k=5, pad=1) -> length 1934 ----
  {
    dim3 g((HW2b + 63) / 64, C2 / 64, Bn);
    conv_gemm_wmma<128, 1, 1936, 1, 5, 0, 1, 1, 1934>
        <<<g, blk, 0, stream>>>(s2, w1_2q, b1_2q, q2, C2);
    conv_gemm_wmma<128, 1, 1936, 1, 5, 0, 1, 1, 1934>
        <<<g, blk, 0, stream>>>(s2, w1_2k, b1_2k, k2, C2);
  }

  // ---- att22 = softmax(q2^T @ r2); r2f = k2 @ att22 ----
  {
    dim3 g((HW2 + 63) / 64, (HW2b + 63) / 64, Bn);
    gemm_tn_wmma<<<g, blk, 0, stream>>>(q2, r2, att22, HW2b, HW2, C2,
                                        (long)C2 * HW2b, (long)C2 * HW2, (long)HW2b * HW2);
    softmax_rows<<<dim3(Bn * HW2b), dim3(256), 0, stream>>>(att22, HW2);
    dim3 g2((HW2 + 63) / 64, C2 / 64, Bn);
    gemm_nn_wmma<<<g2, blk, 0, stream>>>(k2, att22, nullptr, r2f, C2, HW2, HW2b,
                                         (long)C2 * HW2b, (long)HW2b * HW2,
                                         0L, (long)C2 * HW2);
  }

  // ---- out = BN_ReLU(conv3x3(r2f)) + x2 ----
  {
    dim3 g((HW2 + 63) / 64, C2 / 64, Bn);
    conv_gemm_wmma<128, 44, 44, 3, 3, 1, 1, 44, 44>
        <<<g, blk, 0, stream>>>(r2f, w_2r, b_2r, cvo, C2);
    bn_stats<<<dim3(C2), dim3(256), 0, stream>>>(cvo, stats, C2, Bn, HW2);
    long tot = (long)Bn * C2 * HW2;
    bn_relu_apply<<<dim3((unsigned)((tot + 255) / 256)), dim3(256), 0, stream>>>(
        cvo, stats, g_2r, be_2r, x2, out, C2, HW2, tot);
  }
}
